// RiskPropagationGNN_61108794688145
// MI455X (gfx1250) — matrix operand order repaired and checked by reading, not compiled, not run on hardware
//
#include <hip/hip_runtime.h>
#include <math.h>

// ---------------------------------------------------------------------------
// RiskPropagationGNN: 3x edge-aware GAT (heads=1) + LayerNorm + ELU, MLP head.
// N=100000 nodes, E=1600000 edges, D_IN=768, HID=128, EDGE_DIM=7.
// GEMMs via v_wmma_f32_16x16x32_bf16 (bf16 A converted in-register, B pre-converted by convW, f32 acc).
// ---------------------------------------------------------------------------

#define GNN_N   100000
#define GNN_E   1600000
#define GNN_DIN 768
#define GNN_HID 128
#define NEG_SLOPE 0.2f
#define LN_EPS 1e-5f

typedef __attribute__((ext_vector_type(16))) __bf16 v16bf;
typedef __attribute__((ext_vector_type(8)))  __bf16 v8bf;
typedef __attribute__((ext_vector_type(8)))  float  v8f;

// ---------------- small helpers (device) ----------------
__device__ __forceinline__ unsigned enc_f32(float f) {
  unsigned u = __float_as_uint(f);
  return (u & 0x80000000u) ? ~u : (u | 0x80000000u);
}
__device__ __forceinline__ float dec_f32(unsigned e) {
  unsigned u = (e & 0x80000000u) ? (e ^ 0x80000000u) : ~e;
  return __uint_as_float(u);
}
__device__ __forceinline__ float lrelu(float x) { return x > 0.f ? x : NEG_SLOPE * x; }

// ---------------- K0: column sums of edge_attr (for self-loop mean) ----------
__global__ void mea_zero_kernel(float* mea) {
  if (blockIdx.x == 0 && threadIdx.x < 7) mea[threadIdx.x] = 0.f;
}

__global__ void mea_sum_kernel(const float* __restrict__ ea, float* mea, int E) {
  float acc[7] = {0.f, 0.f, 0.f, 0.f, 0.f, 0.f, 0.f};
  int stride = gridDim.x * blockDim.x;
  for (int e = blockIdx.x * blockDim.x + threadIdx.x; e < E; e += stride) {
    const float* p = ea + (size_t)e * 7;
#pragma unroll
    for (int r = 0; r < 7; r++) acc[r] += p[r];
  }
#pragma unroll
  for (int r = 0; r < 7; r++)
    for (int o = 16; o > 0; o >>= 1) acc[r] += __shfl_xor(acc[r], o, 32);
  if ((threadIdx.x & 31) == 0)
#pragma unroll
    for (int r = 0; r < 7; r++) atomicAdd(mea + r, acc[r]);
}

// ---------------- K1: per-layer tiny prep: we[r]=W_edge@att_edge, ae_loop ----
__global__ void prep_kernel(const float* __restrict__ W_edge,
                            const float* __restrict__ att_edge,
                            const float* __restrict__ mea,
                            float* we, float* ae_loop, float invE) {
  __shared__ float sw[7];
  int t = threadIdx.x;
  if (t < 7) {
    float s = 0.f;
    for (int c = 0; c < GNN_HID; c++) s += W_edge[t * GNN_HID + c] * att_edge[c];
    sw[t] = s;
    we[t] = s;
  }
  __syncthreads();
  if (t == 0) {
    float s = 0.f;
    for (int r = 0; r < 7; r++) s += sw[r] * mea[r] * invE;
    *ae_loop = s;
  }
}

// ---------------- K2a: convert+transpose W [K,128] f32 -> Wt [128,K] bf16 ----
__global__ void convW_kernel(const float* __restrict__ W, __bf16* __restrict__ Wt, int K) {
  int i = blockIdx.x * blockDim.x + threadIdx.x;  // over 128*K
  if (i < K * GNN_HID) {
    int n = i / K, k = i % K;
    Wt[i] = (__bf16)W[(size_t)k * GNN_HID + n];
  }
}

// ---------------- K2: WMMA GEMM  H[nrows,128] = A[nrows,K] @ W[K,128] --------
// One wave per 16-row strip; wave keeps all 8 column tiles (8 v8f accumulators)
// so A is read exactly once. A converted f32->bf16 in-register.
__global__ __launch_bounds__(32) void gemm_wmma_kernel(const float* __restrict__ A,
                                                       const __bf16* __restrict__ Bt,
                                                       float* __restrict__ H, int K) {
  const int lane = threadIdx.x;
  const int m0 = blockIdx.x * 16;
  const int nn = lane & 15;
  const bool hi = lane >= 16;
  const int kbA = hi ? 8 : 0;    // A: K-half offset for upper lanes
  const int kbB = hi ? 8 : 0;    // B: K-half offset for upper lanes
  const float* arow = A + (size_t)(m0 + nn) * K;

  v8f acc[8] = {};
  for (int k = 0; k < K; k += 32) {
    v8f a0 = *(const v8f*)(arow + k + kbA);
    v8f a1 = *(const v8f*)(arow + k + 16 + kbA);
    v16bf a;
#pragma unroll
    for (int i = 0; i < 8; i++) { a[i] = (__bf16)a0[i]; a[i + 8] = (__bf16)a1[i]; }
#pragma unroll
    for (int t = 0; t < 8; t++) {
      const __bf16* bp = Bt + (size_t)(t * 16 + nn) * K + k + kbB;
      v8bf b0 = *(const v8bf*)bp;
      v8bf b1 = *(const v8bf*)(bp + 16);
      v16bf b = __builtin_shufflevector(b0, b1, 0, 1, 2, 3, 4, 5, 6, 7,
                                        8, 9, 10, 11, 12, 13, 14, 15);
      acc[t] = __builtin_amdgcn_wmma_f32_16x16x32_bf16(
          false, a, false, b, (short)0, acc[t], false, false);
    }
  }
  const int mr = m0 + (hi ? 8 : 0);
#pragma unroll
  for (int t = 0; t < 8; t++)
#pragma unroll
    for (int r = 0; r < 8; r++)
      H[(size_t)(mr + r) * GNN_HID + t * 16 + nn] = acc[t][r];
}

// ---------------- K3: per-node attention scalars + self-loop max init --------
__global__ void node_att_kernel(const float* __restrict__ h,
                                const float* __restrict__ att_src,
                                const float* __restrict__ att_dst,
                                const float* __restrict__ ae_loop_p,
                                float* a_src, float* a_dst, float* a_loop,
                                unsigned* amax_u, int N) {
  int tid = blockIdx.x * blockDim.x + threadIdx.x;
  int node = tid >> 5, lane = tid & 31;
  if (node >= N) return;
  float4 hv = *(const float4*)(h + (size_t)node * GNN_HID + lane * 4);
  float4 as = *(const float4*)(att_src + lane * 4);
  float4 ad = *(const float4*)(att_dst + lane * 4);
  float s = hv.x * as.x + hv.y * as.y + hv.z * as.z + hv.w * as.w;
  float d = hv.x * ad.x + hv.y * ad.y + hv.z * ad.z + hv.w * ad.w;
#pragma unroll
  for (int o = 16; o > 0; o >>= 1) { s += __shfl_xor(s, o, 32); d += __shfl_xor(d, o, 32); }
  if (lane == 0) {
    a_src[node] = s;
    a_dst[node] = d;
    float al = lrelu(s + d + *ae_loop_p);
    a_loop[node] = al;
    amax_u[node] = enc_f32(al);
  }
}

// ---------------- K4: edge attention term ae[e] = edge_attr[e] . we ----------
__global__ void edge_ae_kernel(const float* __restrict__ ea, const float* __restrict__ we,
                               float* __restrict__ ae, int E) {
  int e = blockIdx.x * blockDim.x + threadIdx.x;
  if (e >= E) return;
  const float* p = ea + (size_t)e * 7;
  float s = 0.f;
#pragma unroll
  for (int r = 0; r < 7; r++) s += p[r] * we[r];
  ae[e] = s;
}

// ---------------- K5: edge logits + segment max (encoded uint atomicMax) -----
__global__ void edge_max_kernel(const int* __restrict__ src, const int* __restrict__ dst,
                                const float* __restrict__ a_src, const float* __restrict__ a_dst,
                                const float* __restrict__ ae, float* __restrict__ alpha,
                                unsigned* amax_u, int E) {
  int e = blockIdx.x * blockDim.x + threadIdx.x;
  if (e >= E) return;
  int s = src[e], d = dst[e];
  float al = lrelu(a_src[s] + a_dst[d] + ae[e]);
  alpha[e] = al;
  atomicMax(amax_u + d, enc_f32(al));
}

// ---------------- K6: decode max, init denom with self-loop term -------------
__global__ void node_max_denom_kernel(const unsigned* __restrict__ amax_u,
                                      const float* __restrict__ a_loop,
                                      float* amax_f, float* denom, int N) {
  int i = blockIdx.x * blockDim.x + threadIdx.x;
  if (i >= N) return;
  float m = dec_f32(amax_u[i]);
  amax_f[i] = m;
  denom[i] = expf(a_loop[i] - m);
}

// ---------------- K7: denom accumulation over edges --------------------------
__global__ void edge_denom_kernel(const int* __restrict__ dst,
                                  const float* __restrict__ alpha,
                                  const float* __restrict__ amax_f,
                                  float* denom, int E) {
  int e = blockIdx.x * blockDim.x + threadIdx.x;
  if (e >= E) return;
  int d = dst[e];
  atomicAdd(denom + d, expf(alpha[e] - amax_f[d]));
}

// ---------------- K8: out init = self-loop contribution ----------------------
__global__ void node_out_init_kernel(const float* __restrict__ h,
                                     const float* __restrict__ a_loop,
                                     const float* __restrict__ amax_f,
                                     const float* __restrict__ denom,
                                     float* __restrict__ out, int N) {
  int q = blockIdx.x * blockDim.x + threadIdx.x;  // N*32 quads
  if (q >= N * 32) return;
  int i = q >> 5, c = (q & 31) * 4;
  float w = expf(a_loop[i] - amax_f[i]) / denom[i];
  float4 hv = *(const float4*)(h + (size_t)i * GNN_HID + c);
  float4 o = {hv.x * w, hv.y * w, hv.z * w, hv.w * w};
  *(float4*)(out + (size_t)i * GNN_HID + c) = o;
}

// ---------------- K9: weighted scatter-add over edges (wave per edge) --------
__global__ void edge_scatter_kernel(const int* __restrict__ src, const int* __restrict__ dst,
                                    const float* __restrict__ alpha,
                                    const float* __restrict__ amax_f,
                                    const float* __restrict__ denom,
                                    const float* __restrict__ h,
                                    float* __restrict__ out, int E) {
  int q = blockIdx.x * blockDim.x + threadIdx.x;  // E*32
  int e = q >> 5, lane = q & 31;
  if (e >= E) return;
  int s = src[e], d = dst[e];
  float w = expf(alpha[e] - amax_f[d]) / denom[d];
  float4 hv = *(const float4*)(h + (size_t)s * GNN_HID + lane * 4);
  float* op = out + (size_t)d * GNN_HID + lane * 4;
  atomicAdd(op + 0, hv.x * w);
  atomicAdd(op + 1, hv.y * w);
  atomicAdd(op + 2, hv.z * w);
  atomicAdd(op + 3, hv.w * w);
}

// ---------------- K10: +bias, ELU, LayerNorm (wave per node) -----------------
__global__ void node_post_kernel(const float* __restrict__ out, const float* __restrict__ bias,
                                 const float* __restrict__ g, const float* __restrict__ bb,
                                 float* __restrict__ xc, int N) {
  int tid = blockIdx.x * blockDim.x + threadIdx.x;
  int node = tid >> 5, lane = tid & 31;
  if (node >= N) return;
  float4 v = *(const float4*)(out + (size_t)node * GNN_HID + lane * 4);
  float4 b = *(const float4*)(bias + lane * 4);
  float x0 = v.x + b.x, x1 = v.y + b.y, x2 = v.z + b.z, x3 = v.w + b.w;
  x0 = x0 > 0.f ? x0 : expf(x0) - 1.f;
  x1 = x1 > 0.f ? x1 : expf(x1) - 1.f;
  x2 = x2 > 0.f ? x2 : expf(x2) - 1.f;
  x3 = x3 > 0.f ? x3 : expf(x3) - 1.f;
  float s = x0 + x1 + x2 + x3;
#pragma unroll
  for (int o = 16; o > 0; o >>= 1) s += __shfl_xor(s, o, 32);
  float mu = s * (1.f / 128.f);
  float d0 = x0 - mu, d1 = x1 - mu, d2 = x2 - mu, d3 = x3 - mu;
  float q = d0 * d0 + d1 * d1 + d2 * d2 + d3 * d3;
#pragma unroll
  for (int o = 16; o > 0; o >>= 1) q += __shfl_xor(q, o, 32);
  float inv = rsqrtf(q * (1.f / 128.f) + LN_EPS);
  float4 gg = *(const float4*)(g + lane * 4);
  float4 bv = *(const float4*)(bb + lane * 4);
  float4 y = {d0 * inv * gg.x + bv.x, d1 * inv * gg.y + bv.y,
              d2 * inv * gg.z + bv.z, d3 * inv * gg.w + bv.w};
  *(float4*)(xc + (size_t)node * GNN_HID + lane * 4) = y;
}

// ---------------- K11: MLP head (4 nodes / 256-thread block, LDS staged) -----
__global__ __launch_bounds__(256) void mlp_kernel(const float* __restrict__ xc,
    const float* __restrict__ W1, const float* __restrict__ b1,
    const float* __restrict__ W2, const float* __restrict__ b2,
    const float* __restrict__ W3, const float* __restrict__ b3,
    float* __restrict__ outp, int N) {
  __shared__ float sx[4][128];
  __shared__ float sh1[4][64];
  __shared__ float sh2[4][32];
  int t = threadIdx.x & 63;
  int slot = threadIdx.x >> 6;
  int node = blockIdx.x * 4 + slot;
  bool valid = node < N;
  if (valid) {
    sx[slot][t] = xc[(size_t)node * 128 + t];
    sx[slot][t + 64] = xc[(size_t)node * 128 + t + 64];
  }
  __syncthreads();
  if (valid) {
    float s = b1[t];
    for (int c = 0; c < 128; c++) s += sx[slot][c] * W1[c * 64 + t];
    sh1[slot][t] = s > 0.f ? s : 0.f;
  }
  __syncthreads();
  if (valid && t < 32) {
    float s = b2[t];
    for (int c = 0; c < 64; c++) s += sh1[slot][c] * W2[c * 32 + t];
    sh2[slot][t] = s > 0.f ? s : 0.f;
  }
  __syncthreads();
  if (valid && t == 0) {
    float s = b3[0];
    for (int c = 0; c < 32; c++) s += sh2[slot][c] * W3[c * 2];
    outp[node] = 1.f / (1.f + expf(-s));
  }
}

// ---------------------------------------------------------------------------
extern "C" void kernel_launch(void* const* d_in, const int* in_sizes, int n_in,
                              void* d_out, int out_size, void* d_ws, size_t ws_size,
                              hipStream_t stream) {
  const int N = GNN_N, E = GNN_E;
  if (n_in < 33) return;

  // ---- detect flatten order: insertion (x first) vs pytree-sorted (edge_attr first)
  bool sortedKeys = (in_sizes[0] != N * GNN_DIN);

  int ix_x, ix_ei, ix_ea, mbase;
  int o_W, o_We, o_asrc, o_adst, o_aedge, o_bias, o_lng, o_lnb, lbase0;
  int m_W1, m_b1, m_W2, m_b2, m_W3, m_b3;
  if (!sortedKeys) {
    ix_x = 0; ix_ei = 1; ix_ea = 2; lbase0 = 3;
    o_W = 0; o_We = 1; o_asrc = 2; o_adst = 3; o_aedge = 4; o_bias = 5; o_lng = 6; o_lnb = 7;
    mbase = 27; m_W1 = 0; m_b1 = 1; m_W2 = 2; m_b2 = 3; m_W3 = 4; m_b3 = 5;
  } else {
    ix_ea = 0; ix_ei = 1; lbase0 = 2; ix_x = 32;
    o_W = 0; o_We = 1; o_adst = 2; o_aedge = 3; o_asrc = 4; o_bias = 5; o_lnb = 6; o_lng = 7;
    mbase = 26; m_W1 = 0; m_W2 = 1; m_W3 = 2; m_b1 = 3; m_b2 = 4; m_b3 = 5;
  }

  const float* x  = (const float*)d_in[ix_x];
  const int*   ei = (const int*)d_in[ix_ei];
  const float* ea = (const float*)d_in[ix_ea];
  const int* src = ei;
  const int* dst = ei + E;

  const float *LW[3], *LWe[3], *Lasrc[3], *Ladst[3], *Laedge[3], *Lbias[3], *Llng[3], *Llnb[3];
  for (int l = 0; l < 3; l++) {
    int b = lbase0 + 8 * l;
    LW[l]     = (const float*)d_in[b + o_W];
    LWe[l]    = (const float*)d_in[b + o_We];
    Lasrc[l]  = (const float*)d_in[b + o_asrc];
    Ladst[l]  = (const float*)d_in[b + o_adst];
    Laedge[l] = (const float*)d_in[b + o_aedge];
    Lbias[l]  = (const float*)d_in[b + o_bias];
    Llng[l]   = (const float*)d_in[b + o_lng];
    Llnb[l]   = (const float*)d_in[b + o_lnb];
  }
  const float* W1 = (const float*)d_in[mbase + m_W1];
  const float* b1 = (const float*)d_in[mbase + m_b1];
  const float* W2 = (const float*)d_in[mbase + m_W2];
  const float* b2 = (const float*)d_in[mbase + m_b2];
  const float* W3 = (const float*)d_in[mbase + m_W3];
  const float* b3 = (const float*)d_in[mbase + m_b3];

  // ---- carve workspace (256-byte aligned blocks) ----
  size_t off = 0;
  char* base = (char*)d_ws;
  auto carve = [&](size_t bytes) -> void* {
    void* p = base + off;
    off = (off + bytes + 255) & ~(size_t)255;
    return p;
  };
  float*    h       = (float*)carve((size_t)N * GNN_HID * 4);
  float*    out     = (float*)carve((size_t)N * GNN_HID * 4);
  float*    xc      = (float*)carve((size_t)N * GNN_HID * 4);
  float*    aeArr   = (float*)carve((size_t)E * 4);
  float*    alpha   = (float*)carve((size_t)E * 4);
  float*    a_src   = (float*)carve((size_t)N * 4);
  float*    a_dst   = (float*)carve((size_t)N * 4);
  float*    a_loop  = (float*)carve((size_t)N * 4);
  unsigned* amax_u  = (unsigned*)carve((size_t)N * 4);
  float*    amax_f  = (float*)carve((size_t)N * 4);
  float*    denom   = (float*)carve((size_t)N * 4);
  __bf16*   Wt      = (__bf16*)carve((size_t)GNN_DIN * GNN_HID * 2);
  float*    mea     = (float*)carve(64);
  float*    we      = (float*)carve(64);
  float*    ae_loop = (float*)carve(64);
  if (off > ws_size) return;  // workspace insufficient; bail safely

  // ---- K0: self-loop edge_attr mean (column sums; divided by E in prep) ----
  mea_zero_kernel<<<1, 32, 0, stream>>>(mea);
  mea_sum_kernel<<<512, 256, 0, stream>>>(ea, mea, E);

  const float* xin = x;
  int K = GNN_DIN;
  for (int l = 0; l < 3; l++) {
    prep_kernel<<<1, 32, 0, stream>>>(LWe[l], Laedge[l], mea, we, ae_loop, 1.0f / (float)E);
    convW_kernel<<<(K * GNN_HID + 255) / 256, 256, 0, stream>>>(LW[l], Wt, K);
    gemm_wmma_kernel<<<N / 16, 32, 0, stream>>>(xin, Wt, h, K);
    node_att_kernel<<<(N * 32 + 255) / 256, 256, 0, stream>>>(
        h, Lasrc[l], Ladst[l], ae_loop, a_src, a_dst, a_loop, amax_u, N);
    edge_ae_kernel<<<(E + 255) / 256, 256, 0, stream>>>(ea, we, aeArr, E);
    edge_max_kernel<<<(E + 255) / 256, 256, 0, stream>>>(
        src, dst, a_src, a_dst, aeArr, alpha, amax_u, E);
    node_max_denom_kernel<<<(N + 255) / 256, 256, 0, stream>>>(amax_u, a_loop, amax_f, denom, N);
    edge_denom_kernel<<<(E + 255) / 256, 256, 0, stream>>>(dst, alpha, amax_f, denom, E);
    node_out_init_kernel<<<(N * 32 + 255) / 256, 256, 0, stream>>>(
        h, a_loop, amax_f, denom, out, N);
    edge_scatter_kernel<<<(E * 32 + 255) / 256, 256, 0, stream>>>(
        src, dst, alpha, amax_f, denom, h, out, E);
    node_post_kernel<<<(N * 32 + 255) / 256, 256, 0, stream>>>(
        out, Lbias[l], Llng[l], Llnb[l], xc, N);
    xin = xc;
    K = GNN_HID;
  }

  mlp_kernel<<<(N + 3) / 4, 256, 0, stream>>>(xc, W1, b1, W2, b2, W3, b3, (float*)d_out, N);
}
